// Hippocampus_69939247448534
// MI455X (gfx1250) — compile-verified, matrix operand order checked by reading, not run on hardware
//
#include <hip/hip_runtime.h>

// ---------------------------------------------------------------------------
// Hippocampus memory-attention, fused for MI455X (gfx1250, wave32, WMMA bf16)
//   q     = x @ W^T + b          [16384,1024]
//   attn  = (q @ mem^T) / 32     [16384,256]
//   s     = softmax(attn)        [16384,256]
//   out   = s @ mem              [16384,1024]
// Block = 64 rows, 8 waves. Register tiling: 4 row-tiles share each B frag.
// x staged through LDS by the Tensor Data Mover (TENSOR_LOAD_TO_LDS).
// ---------------------------------------------------------------------------

typedef __attribute__((ext_vector_type(16))) __bf16 v16bf;
typedef __attribute__((ext_vector_type(2)))  __bf16 v2bf;
typedef __attribute__((ext_vector_type(8)))  float  v8f;
typedef __attribute__((ext_vector_type(2)))  float  v2f;
typedef __attribute__((ext_vector_type(4)))  unsigned int v4u;
typedef __attribute__((ext_vector_type(8)))  int    v8i;
typedef __attribute__((ext_vector_type(4)))  int    v4i;

union FragBF {
    v16bf bf;
    uint4 u[2];
};

// native f32 -> bf16 (RNE) conversions; backend selects v_cvt_pk_bf16_f32
__device__ __forceinline__ unsigned short f2bf(float f) {
    return __builtin_bit_cast(unsigned short, (__bf16)f);
}
__device__ __forceinline__ unsigned int pack_bf2(float lo, float hi) {
    v2f  vf = {lo, hi};
    v2bf vb = __builtin_convertvector(vf, v2bf);
    return __builtin_bit_cast(unsigned int, vb);
}
__device__ __forceinline__ unsigned lds_off(const void* p) {
    // generic LDS address: low 32 bits are the LDS byte address (ISA §10.2 aperture)
    return (unsigned)(unsigned long long)(uintptr_t)p;
}

// ------------------------------ constants ----------------------------------
constexpr int DIM   = 1024;   // feature dim
constexpr int MEMN  = 256;    // memory slots
constexpr int NROW  = 16384;  // B*T
constexpr int RPB   = 64;     // rows per block
constexpr float SCALE = 0.03125f;  // 1/sqrt(1024)

// workspace layout (bytes)
constexpr size_t WS_WBF   = 0;                               // 1024*1024 bf16 = 2 MB
constexpr size_t WS_MEMBF = 2u * 1024u * 1024u;              // 256*1024 bf16 = 512 KB
constexpr size_t WS_MEMT  = WS_MEMBF + 512u * 1024u;         // 1024*256 bf16 = 512 KB

// LDS layout (bytes), total 256 KB dynamic
constexpr unsigned L_XBF    = 0;        // 64x1024 bf16 = 128 KB
constexpr unsigned L_QBF    = 131072;   // 64x1024 bf16 = 128 KB (also TDM fp32 staging)
constexpr unsigned L_ATTN   = 0;        // 64x256 f32 = 64 KB   (overlay on XBF)
constexpr unsigned L_SCORES = 65536;    // 64x256 bf16 = 32 KB
constexpr unsigned L_RMAX   = 98304;    // 64x4 f32 = 1 KB
constexpr unsigned L_RSUM   = 99328;    // 64x4 f32 = 1 KB

// ---------------------------------------------------------------------------
// Prep: fp32 -> bf16 conversions (W, mem) and mem transpose (memT[d][m]).
// ---------------------------------------------------------------------------
__global__ void hippo_prep(const float* __restrict__ W,
                           const float* __restrict__ mem,
                           unsigned int* __restrict__ wbf,
                           unsigned int* __restrict__ membf,
                           unsigned int* __restrict__ memT)
{
    const int NW = (DIM * DIM) / 2;
    const int NM = (MEMN * DIM) / 2;
    const int NT = (DIM * MEMN) / 2;
    int p = blockIdx.x * blockDim.x + threadIdx.x;
    if (p < NW) {
        wbf[p] = pack_bf2(W[2 * p], W[2 * p + 1]);
    } else if (p < NW + NM) {
        int q = p - NW;
        membf[q] = pack_bf2(mem[2 * q], mem[2 * q + 1]);
    } else if (p < NW + NM + NT) {
        int q = p - NW - NM;
        int d  = q >> 7;            // feature index (row of memT)
        int m2 = (q & 127) * 2;     // memory index pair
        memT[q] = pack_bf2(mem[m2 * DIM + d], mem[(m2 + 1) * DIM + d]);
    }
}

// ---------------------------------------------------------------------------
// TDM: async-DMA a (rows x 1024) fp32 tile of x into LDS.
// D# per CDNA5 ISA; 6-arg builtin form on this toolchain:
//   (uint32x4 g0, int32x8 g1, int32x4 g2, int32x4 g3, int32x8 g4, i32 cpol)
// ---------------------------------------------------------------------------
__device__ __forceinline__ void tdm_load_x_tile(const float* gsrc, unsigned lds_byte,
                                                int rows) {
    unsigned long long ga = (unsigned long long)(uintptr_t)gsrc;
    v4u g0;
    g0[0] = 1u;                                            // count=1, user mode
    g0[1] = lds_byte;                                      // lds_addr
    g0[2] = (unsigned)(ga & 0xffffffffu);                  // global_addr[31:0]
    g0[3] = (unsigned)((ga >> 32) & 0x1ffffffu)            // global_addr[56:32]
          | (2u << 30);                                    // type = 2 ("image")
    v8i g1;
    g1[0] = 0x20000;                                       // data_size = 2 (4 bytes)
    g1[1] = (DIM  << 16);                                  // tensor_dim0[15:0] @ bits 63:48
    g1[2] = (NROW << 16);                                  // tensor_dim1[15:0] @ bits 95:80
    g1[3] = (DIM  << 16);                                  // tile_dim0 @ bits 127:112
    g1[4] = rows;                                          // tile_dim1 @ bits 143:128
    g1[5] = DIM;                                           // tensor_dim0_stride[31:0]
    g1[6] = 0;
    g1[7] = 0;
    v4i gz4 = {0, 0, 0, 0};
    v8i gz8 = {0, 0, 0, 0, 0, 0, 0, 0};
    __builtin_amdgcn_tensor_load_to_lds(g0, g1, gz4, gz4, gz8, 0);
}

// ---------------------------------------------------------------------------
// Fused main kernel: one block = 64 rows, 8 waves, 256 threads.
// ---------------------------------------------------------------------------
__global__ __launch_bounds__(256)
void hippo_main(const float* __restrict__ x,
                const float* __restrict__ bias,
                const unsigned short* __restrict__ wbf,    // [DIM][DIM] bf16
                const unsigned short* __restrict__ membf,  // [MEMN][DIM] bf16
                const unsigned short* __restrict__ memTbf, // [DIM][MEMN] bf16
                float* __restrict__ out)
{
    extern __shared__ char smem[];
    unsigned short* xs     = (unsigned short*)(smem + L_XBF);
    unsigned short* qs     = (unsigned short*)(smem + L_QBF);
    float*          attnf  = (float*)(smem + L_ATTN);
    unsigned short* scores = (unsigned short*)(smem + L_SCORES);
    float*          redmax = (float*)(smem + L_RMAX);
    float*          redsum = (float*)(smem + L_RSUM);

    const int tid  = threadIdx.x;
    const int wave = tid >> 5;
    const int lane = tid & 31;
    const int half = lane >> 4;   // which 16-lane half (ISA A/B/C layouts)
    const int l    = lane & 15;
    const int row0 = blockIdx.x * RPB;

    // -------- stage x (64x1024 fp32) -> bf16 LDS via TDM, 2 chunks ---------
    for (int c = 0; c < 2; ++c) {
        if (wave == 0) {
            tdm_load_x_tile(x + (size_t)(row0 + c * 32) * DIM, lds_off(smem + L_QBF), 32);
            __builtin_amdgcn_s_wait_tensorcnt(0);
        }
        __syncthreads();
        const float4*  src = (const float4*)(smem + L_QBF);
        uint2*         dst = (uint2*)((unsigned int*)xs + c * 32 * (DIM / 2));
        #pragma unroll 4
        for (int i = 0; i < 32; ++i) {           // 8192 fp32-quads / 256 threads
            int e = tid + 256 * i;
            float4 v = src[e];
            uint2 pk;
            pk.x = pack_bf2(v.x, v.y);
            pk.y = pack_bf2(v.z, v.w);
            dst[e] = pk;
        }
        __syncthreads();
    }

    // -------- q = x @ W^T + b : 8 n-tiles/wave x 4 row-tiles, K = 1024 -----
    for (int i = 0; i < 8; ++i) {
        const int n0 = (wave + 8 * i) * 16;
        v8f acc[4] = {{}, {}, {}, {}};
        const unsigned short* wrow = wbf + (size_t)(n0 + l) * DIM;   // B column
        #pragma unroll 2
        for (int k = 0; k < DIM; k += 32) {
            const int kb = k + half * 8;
            // warm next k-chunk of the B stream (speculative, 1 KB ahead)
            __builtin_prefetch(wrow + kb + 512, 0, 0);
            FragBF b;
            b.u[0] = *(const uint4*)(wrow + kb);
            b.u[1] = *(const uint4*)(wrow + kb + 16);
            #pragma unroll
            for (int rt = 0; rt < 4; ++rt) {
                FragBF a;
                const unsigned short* xrow = xs + (rt * 16 + l) * DIM;
                a.u[0] = *(const uint4*)(xrow + kb);
                a.u[1] = *(const uint4*)(xrow + kb + 16);
                acc[rt] = __builtin_amdgcn_wmma_f32_16x16x32_bf16(
                    false, a.bf, false, b.bf, (short)0, acc[rt], false, false);
            }
        }
        const float bv = bias[n0 + l];
        #pragma unroll
        for (int rt = 0; rt < 4; ++rt)
            #pragma unroll
            for (int r = 0; r < 8; ++r)
                qs[(rt * 16 + r + half * 8) * DIM + n0 + l] = f2bf(acc[rt][r] + bv);
    }
    __syncthreads();   // q complete; x region becomes free

    // -------- attn = (q @ mem^T) * scale : 2 m-tiles/wave x 4 row-tiles ----
    for (int i = 0; i < 2; ++i) {
        const int m0 = (wave * 2 + i) * 16;
        v8f acc[4] = {{}, {}, {}, {}};
        const unsigned short* mrow = membf + (size_t)(m0 + l) * DIM;
        #pragma unroll 2
        for (int k = 0; k < DIM; k += 32) {
            const int kb = k + half * 8;
            FragBF b;
            b.u[0] = *(const uint4*)(mrow + kb);
            b.u[1] = *(const uint4*)(mrow + kb + 16);
            #pragma unroll
            for (int rt = 0; rt < 4; ++rt) {
                FragBF a;
                const unsigned short* qrow = qs + (rt * 16 + l) * DIM;
                a.u[0] = *(const uint4*)(qrow + kb);
                a.u[1] = *(const uint4*)(qrow + kb + 16);
                acc[rt] = __builtin_amdgcn_wmma_f32_16x16x32_bf16(
                    false, a.bf, false, b.bf, (short)0, acc[rt], false, false);
            }
        }
        #pragma unroll
        for (int rt = 0; rt < 4; ++rt)
            #pragma unroll
            for (int r = 0; r < 8; ++r)
                attnf[(rt * 16 + r + half * 8) * MEMN + m0 + l] = acc[rt][r] * SCALE;
    }
    __syncthreads();

    // -------- row softmax over 256 (4 threads per row, 64 cols each) -------
    {
        const int r = tid >> 2;   // 0..63
        const int j = tid & 3;    // 0..3
        float* rowp = attnf + r * MEMN + j * 64;
        float m = -3.0e38f;
        for (int c = 0; c < 64; ++c) m = fmaxf(m, rowp[c]);
        redmax[r * 4 + j] = m;
        __syncthreads();
        float rm = fmaxf(fmaxf(redmax[r * 4 + 0], redmax[r * 4 + 1]),
                         fmaxf(redmax[r * 4 + 2], redmax[r * 4 + 3]));
        float s = 0.0f;
        for (int c = 0; c < 64; ++c) {
            float e = __expf(rowp[c] - rm);
            rowp[c] = e;
            s += e;
        }
        redsum[r * 4 + j] = s;
        __syncthreads();
        const float tot = redsum[r * 4 + 0] + redsum[r * 4 + 1] +
                          redsum[r * 4 + 2] + redsum[r * 4 + 3];
        const float inv = 1.0f / tot;
        unsigned short* srow = scores + r * MEMN + j * 64;
        for (int c = 0; c < 64; ++c) srow[c] = f2bf(rowp[c] * inv);
    }
    __syncthreads();

    // -------- out = scores @ mem : 8 n-tiles/wave x 4 row-tiles, K = 256 ---
    for (int i = 0; i < 8; ++i) {
        const int n0 = (wave + 8 * i) * 16;
        v8f acc[4] = {{}, {}, {}, {}};
        const unsigned short* trow = memTbf + (size_t)(n0 + l) * MEMN;
        #pragma unroll 2
        for (int k = 0; k < MEMN; k += 32) {
            const int kb = k + half * 8;
            FragBF b;
            b.u[0] = *(const uint4*)(trow + kb);
            b.u[1] = *(const uint4*)(trow + kb + 16);
            #pragma unroll
            for (int rt = 0; rt < 4; ++rt) {
                FragBF a;
                const unsigned short* srow = scores + (rt * 16 + l) * MEMN;
                a.u[0] = *(const uint4*)(srow + kb);
                a.u[1] = *(const uint4*)(srow + kb + 16);
                acc[rt] = __builtin_amdgcn_wmma_f32_16x16x32_bf16(
                    false, a.bf, false, b.bf, (short)0, acc[rt], false, false);
            }
        }
        #pragma unroll
        for (int rt = 0; rt < 4; ++rt)
            #pragma unroll
            for (int r = 0; r < 8; ++r)
                out[(size_t)(row0 + rt * 16 + r + half * 8) * DIM + n0 + l] = acc[rt][r];
    }
}

// ---------------------------------------------------------------------------
extern "C" void kernel_launch(void* const* d_in, const int* in_sizes, int n_in,
                              void* d_out, int out_size, void* d_ws, size_t ws_size,
                              hipStream_t stream) {
    const float* x   = (const float*)d_in[0];   // [4,4096,1024]
    const float* W   = (const float*)d_in[1];   // [1024,1024]
    const float* b   = (const float*)d_in[2];   // [1024]
    const float* mem = (const float*)d_in[3];   // [256,1024]

    char* ws = (char*)d_ws;
    unsigned int* wbf   = (unsigned int*)(ws + WS_WBF);
    unsigned int* membf = (unsigned int*)(ws + WS_MEMBF);
    unsigned int* memT  = (unsigned int*)(ws + WS_MEMT);

    const int totalPairs = (DIM * DIM + MEMN * DIM + DIM * MEMN) / 2;
    hippo_prep<<<(totalPairs + 255) / 256, 256, 0, stream>>>(W, mem, wbf, membf, memT);

    hippo_main<<<NROW / RPB, 256, 262144, stream>>>(
        x, b,
        (const unsigned short*)wbf,
        (const unsigned short*)membf,
        (const unsigned short*)memT,
        (float*)d_out);
}
